// DrugEncoder_2310692405715
// MI455X (gfx1250) — matrix operand order repaired and checked by reading, not compiled, not run on hardware
//
#include <hip/hip_runtime.h>
#include <math.h>

// ---------------------------------------------------------------------------
// MPNN DrugEncoder for gfx1250 (MI455X).
//  - All GEMMs: v_wmma_f32_16x16x32_f16.
//  - Weights pre-packed once per launch into f16 fragment-major layout
//    (fragment = 32 lanes x 32B contiguous).
//  - Edge kernel: packed weights staged into LDS per workgroup via the
//    Tensor Data Mover (TENSOR_LOAD_TO_LDS + s_wait_tensorcnt), 16 edge
//    tiles per wave amortize the staging; B-frags read as ds_load_b128.
//  - f16 mirror of node features (h16) halves the edge-gather HBM traffic.
//  - GRU+BN+residual fused per 16-node tile (gate pairs fused into one acc).
// ---------------------------------------------------------------------------

typedef __attribute__((ext_vector_type(16))) _Float16 v16h;
typedef __attribute__((ext_vector_type(8)))  _Float16 v8h;
typedef __attribute__((ext_vector_type(8)))  float    v8f;
typedef unsigned int u32x4 __attribute__((ext_vector_type(4)));
typedef int          i32x8 __attribute__((ext_vector_type(8)));
typedef int          i32x4 __attribute__((ext_vector_type(4)));

#define HID      128
#define EDIM     16
#define NNODES   50000
#define NEDGES   800000
#define NGRAPHS  256
#define BN_EPS   1e-5f
#define FRAG     512   // halfs per packed 32x16 B-fragment (32 lanes x 16)

__device__ __forceinline__ v8f wmma16(v16h a, v16h b, v8f c) {
  return __builtin_amdgcn_wmma_f32_16x16x32_f16(false, a, false, b, (short)0, c,
                                                false, false);
}

// Packed B fragment from global: lane L reads 32 contiguous bytes (2x b128).
__device__ __forceinline__ v16h b_packed(const _Float16* __restrict__ P, int f,
                                         int lane) {
  return *(const v16h*)(P + (((size_t)f * 32 + lane) << 4));
}

// Packed B fragment from LDS: 2x ds_load_b128.
__device__ __forceinline__ v16h b_lds(const _Float16* S, int f, int lane) {
  return *(const v16h*)(S + (((unsigned)f * 32 + (unsigned)lane) << 4));
}

// A fragment (16x32 f16) from an f16 row: two aligned 16B vector loads.
__device__ __forceinline__ v16h a_frag_h(const _Float16* __restrict__ row, int kb,
                                         int kh) {
  v8h lo = *(const v8h*)(row + kb + 8 * kh);
  v8h hi = *(const v8h*)(row + kb + 16 + 8 * kh);
  v16h a;
#pragma unroll
  for (int i = 0; i < 8; ++i) { a[i] = lo[i]; a[i + 8] = hi[i]; }
  return a;
}

// A fragment from an f32 row (agg / x / zcat), cvt on the fly.
__device__ __forceinline__ v16h a_frag_f(const float* __restrict__ row, int kb,
                                         int kh) {
  v16h a;
#pragma unroll
  for (int i = 0; i < 8; ++i) {
    a[i]     = (_Float16)row[kb + 8 * kh + i];
    a[i + 8] = (_Float16)row[kb + 16 + 8 * kh + i];
  }
  return a;
}

__device__ __forceinline__ float sigm(float x) { return 1.0f / (1.0f + __expf(-x)); }

// monotonic uint key for float atomic-max
__device__ __forceinline__ unsigned fkey(float x) {
  unsigned u = __float_as_uint(x);
  return (u & 0x80000000u) ? ~u : (u | 0x80000000u);
}
__device__ __forceinline__ float funkey(unsigned k) {
  return __uint_as_float((k & 0x80000000u) ? (k & 0x7FFFFFFFu) : ~k);
}

// ---------------------------------------------------------------------------
// TDM: 1-D contiguous global -> LDS copy of nDwords DWORDs.
// D# per CDNA5 ISA:
//  g0: count=1 | lds_addr | global_addr[56:0] | type=2
//  g1: data_size=4B, tensor_dim0=tile_dim0=nDwords, tensor_dim1=1,
//      tensor_dim0_stride=nDwords.  groups 2/3 zero (<=2D tensor).
// amdgpu-toolchain (clang-23) 6-arg builtin form.
// ---------------------------------------------------------------------------
__device__ __forceinline__ void tdm_load_1d(unsigned lds_addr, const void* gptr,
                                            unsigned nDwords) {
  unsigned long long ga = (unsigned long long)gptr;
  u32x4 g0;
  g0[0] = 1u;                                   // count=1 (valid user D#)
  g0[1] = lds_addr;                             // LDS byte address
  g0[2] = (unsigned)(ga & 0xFFFFFFFFu);         // global_addr[31:0]
  g0[3] = (unsigned)((ga >> 32) & 0x01FFFFFFu)  // global_addr[56:32]
          | (2u << 30);                         // type=2 ("image")
  i32x8 g1;
  g1[0] = (int)(2u << 16);                      // data_size=2 -> 4 bytes
  g1[1] = (int)((nDwords & 0xFFFFu) << 16);     // tensor_dim0[15:0]
  g1[2] = (int)(((nDwords >> 16) & 0xFFFFu)     // tensor_dim0[31:16]
          | (1u << 16));                        // tensor_dim1 = 1
  g1[3] = (int)((nDwords & 0xFFFFu) << 16);     // tile_dim0 = nDwords
  g1[4] = 0;                                    // tile_dim1/2 unused
  g1[5] = (int)nDwords;                         // tensor_dim0_stride lo
  g1[6] = 0;
  g1[7] = 0;
  i32x4 z4 = {0, 0, 0, 0};
  i32x8 z8 = {0, 0, 0, 0, 0, 0, 0, 0};
  __builtin_amdgcn_tensor_load_to_lds(g0, g1, z4, z4, z8, 0);
}

// ---------------------------------------------------------------------------
// Weight packers (run once per launch; ~1 MB total).
// ---------------------------------------------------------------------------
__global__ void k_pack_km(const float* __restrict__ W, int ldn, int nKT,
                          _Float16* __restrict__ out, int total) {
  int idx = blockIdx.x * blockDim.x + threadIdx.x;
  if (idx >= total) return;
  int f = idx >> 9, r = idx & 511, L = r >> 4, i = r & 15;
  int nt = f / nKT, kt = f - nt * nKT;
  int kh = L >> 4, n = L & 15;
  out[idx] = (_Float16)W[(size_t)(kt * 32 + 8 * kh + (i & 7) + 2 * (i & 8)) * ldn + nt * 16 + n];
}

__global__ void k_pack_edge(const float* __restrict__ W1,
                            _Float16* __restrict__ out) {
  int idx = blockIdx.x * blockDim.x + threadIdx.x;
  if (idx >= 8 * FRAG) return;
  int f = idx >> 9, r = idx & 511, L = r >> 4, i = r & 15;
  int kh = L >> 4, n = L & 15;
  float v = (i < 8) ? W1[(size_t)(HID + 8 * kh + i) * HID + f * 16 + n] : 0.0f;
  out[idx] = (_Float16)v;
}

__global__ void k_pack_tr(const float* __restrict__ W, _Float16* __restrict__ out,
                          int total) {
  int idx = blockIdx.x * blockDim.x + threadIdx.x;
  if (idx >= total) return;
  int f = idx >> 9, r = idx & 511, L = r >> 4, i = r & 15;
  int jt = f >> 2, kt = f & 3;
  int kh = L >> 4, n = L & 15;
  out[idx] = (_Float16)W[(size_t)(jt * 16 + n) * HID + kt * 32 + 8 * kh + (i & 7) + 2 * (i & 8)];
}

// ---------------------------------------------------------------------------
// h = relu(x @ Win + b) ; also writes f16 mirror h16.
// ---------------------------------------------------------------------------
__global__ void k_lin_in(const float* __restrict__ x,
                         const _Float16* __restrict__ PW,
                         const float* __restrict__ bias, float* __restrict__ h,
                         _Float16* __restrict__ h16) {
  const int lane = threadIdx.x & 31, wid = threadIdx.x >> 5;
  const int tile = blockIdx.x * (blockDim.x >> 5) + wid;
  if (tile * 16 >= NNODES) return;
  const int kh = lane >> 4, ln = lane & 15;

  const float* xrow = x + (size_t)(tile * 16 + ln) * HID;
  v16h A[4];
#pragma unroll
  for (int kk = 0; kk < 4; ++kk) A[kk] = a_frag_f(xrow, kk * 32, kh);

#pragma unroll
  for (int nt = 0; nt < 8; ++nt) {
    v16h B0 = b_packed(PW, nt * 4 + 0, lane), B1 = b_packed(PW, nt * 4 + 1, lane);
    v16h B2 = b_packed(PW, nt * 4 + 2, lane), B3 = b_packed(PW, nt * 4 + 3, lane);
    v8f acc = {};
    acc = wmma16(A[0], B0, acc);
    acc = wmma16(A[1], B1, acc);
    acc = wmma16(A[2], B2, acc);
    acc = wmma16(A[3], B3, acc);
    const float bv = bias[nt * 16 + ln];
#pragma unroll
    for (int v = 0; v < 8; ++v) {
      const size_t idx = (size_t)(tile * 16 + v + 8 * kh) * HID + nt * 16 + ln;
      const float val = fmaxf(acc[v] + bv, 0.0f);
      h[idx] = val;
      h16[idx] = (_Float16)val;
    }
  }
}

// ---------------------------------------------------------------------------
// Fused edge MLP + scatter-add.  Weights TDM-staged into LDS per block;
// each wave processes MSG_TPW 16-edge tiles.
// ---------------------------------------------------------------------------
#define MSG_WAVES 8
#define MSG_TPW   16
__global__ void k_msg(const _Float16* __restrict__ h16, const int* __restrict__ ei,
                      const float* __restrict__ ea,
                      const _Float16* __restrict__ P1,  // 40 frags (32 h + 8 edge)
                      const float* __restrict__ b1,
                      const _Float16* __restrict__ P2,  // 32 frags
                      const float* __restrict__ b2, float* __restrict__ agg) {
  __shared__ _Float16 sP1[40 * FRAG];              // 40 KB
  __shared__ _Float16 sP2[32 * FRAG];              // 32 KB
  __shared__ _Float16 m1[MSG_WAVES][16][HID];      // 32 KB
  const int lane = threadIdx.x & 31, wid = threadIdx.x >> 5;

  // Stage packed weights into LDS via the Tensor Data Mover (EXEC-independent,
  // issued by one lane; TENSORcnt tracks completion).
  if (threadIdx.x == 0) {
    tdm_load_1d((unsigned)(unsigned long long)(const void*)sP1, P1,
                (40 * FRAG) / 2);
    tdm_load_1d((unsigned)(unsigned long long)(const void*)sP2, P2,
                (32 * FRAG) / 2);
    __builtin_amdgcn_s_wait_tensorcnt(0);
  }
  __syncthreads();

  const int kh = lane >> 4, ln = lane & 15;
  const int* src = ei;
  const int* dst = ei + NEDGES;
  const int tile0 = blockIdx.x * (MSG_WAVES * MSG_TPW) + wid * MSG_TPW;

  for (int t = 0; t < MSG_TPW; ++t) {
    const int tile = tile0 + t;
    const int e0 = tile * 16;
    if (e0 >= NEDGES) break;  // wave-uniform

    // A: gathered f16 node features + zero-padded edge features
    const int s = src[e0 + ln];
    const _Float16* hrow = h16 + (size_t)s * HID;
    v16h Ah[4];
#pragma unroll
    for (int kk = 0; kk < 4; ++kk) Ah[kk] = a_frag_h(hrow, kk * 32, kh);

    const float* erow = ea + (size_t)(e0 + ln) * EDIM;
    v16h Ae;
#pragma unroll
    for (int i = 0; i < 8; ++i) {
      Ae[i]     = (_Float16)erow[8 * kh + i];  // K = 0..15 valid
      Ae[i + 8] = (_Float16)0.0f;              // K = 16..31 zero pad
    }

    int didx[8];
#pragma unroll
    for (int v = 0; v < 8; ++v) didx[v] = dst[e0 + v + 8 * kh];

    // ---- GEMM1: [16 x 144] @ [144 x 128] + relu -> m1 (f16, wave-private) ----
#pragma unroll
    for (int nt = 0; nt < 8; ++nt) {
      v16h B0 = b_lds(sP1, nt * 4 + 0, lane), B1 = b_lds(sP1, nt * 4 + 1, lane);
      v16h B2 = b_lds(sP1, nt * 4 + 2, lane), B3 = b_lds(sP1, nt * 4 + 3, lane);
      v16h Be = b_lds(sP1, 32 + nt, lane);
      v8f acc = {};
      acc = wmma16(Ah[0], B0, acc);
      acc = wmma16(Ah[1], B1, acc);
      acc = wmma16(Ah[2], B2, acc);
      acc = wmma16(Ah[3], B3, acc);
      acc = wmma16(Ae, Be, acc);
      const float bv = b1[nt * 16 + ln];
#pragma unroll
      for (int v = 0; v < 8; ++v)
        m1[wid][v + 8 * kh][nt * 16 + ln] = (_Float16)fmaxf(acc[v] + bv, 0.0f);
    }
    // m1 is wave-private and DS ops are in-order per wave; only block
    // compiler reordering of the cross-lane LDS accesses.
    asm volatile("" ::: "memory");

    // ---- GEMM2: [16 x 128] @ [128 x 128] -> atomic scatter ----
    v16h Am[4];
#pragma unroll
    for (int kk = 0; kk < 4; ++kk) {
      v8h lo = *(const v8h*)(&m1[wid][ln][kk * 32 + 8 * kh]);
      v8h hi = *(const v8h*)(&m1[wid][ln][kk * 32 + 16 + 8 * kh]);
#pragma unroll
      for (int i = 0; i < 8; ++i) { Am[kk][i] = lo[i]; Am[kk][i + 8] = hi[i]; }
    }
#pragma unroll
    for (int nt = 0; nt < 8; ++nt) {
      v16h B0 = b_lds(sP2, nt * 4 + 0, lane), B1 = b_lds(sP2, nt * 4 + 1, lane);
      v16h B2 = b_lds(sP2, nt * 4 + 2, lane), B3 = b_lds(sP2, nt * 4 + 3, lane);
      v8f acc = {};
      acc = wmma16(Am[0], B0, acc);
      acc = wmma16(Am[1], B1, acc);
      acc = wmma16(Am[2], B2, acc);
      acc = wmma16(Am[3], B3, acc);
      const float bv = b2[nt * 16 + ln];
#pragma unroll
      for (int v = 0; v < 8; ++v)
        atomicAdd(&agg[(size_t)didx[v] * HID + nt * 16 + ln], acc[v] + bv);
    }
    asm volatile("" ::: "memory");  // keep next iter's m1 writes after reads
  }
}

// ---------------------------------------------------------------------------
// Fused GRU + BN + residual, in-place on h / h16.
// ---------------------------------------------------------------------------
__global__ void k_gru(float* __restrict__ h, _Float16* __restrict__ h16,
                      const float* __restrict__ agg,
                      const _Float16* __restrict__ Pih,  // 24 jt x 4 kt frags
                      const _Float16* __restrict__ Phh,
                      const float* __restrict__ bih, const float* __restrict__ bhh,
                      const float* __restrict__ bn_g, const float* __restrict__ bn_b,
                      const float* __restrict__ bn_m, const float* __restrict__ bn_v) {
  const int lane = threadIdx.x & 31, wid = threadIdx.x >> 5;
  const int tile = blockIdx.x * (blockDim.x >> 5) + wid;
  if (tile * 16 >= NNODES) return;
  const int kh = lane >> 4, ln = lane & 15;

  const float* arow = agg + (size_t)(tile * 16 + ln) * HID;
  const _Float16* hrow = h16 + (size_t)(tile * 16 + ln) * HID;
  v16h Aa[4], Ahh[4];
#pragma unroll
  for (int kk = 0; kk < 4; ++kk) {
    Aa[kk]  = a_frag_f(arow, kk * 32, kh);
    Ahh[kk] = a_frag_h(hrow, kk * 32, kh);
  }

#pragma unroll 1
  for (int c = 0; c < 8; ++c) {
    const int j = c * 16 + ln;
    v8f aR = {}, aZ = {}, aN1 = {}, aN2 = {};
#pragma unroll
    for (int kk = 0; kk < 4; ++kk) {  // r gate: ir + hr fused
      aR = wmma16(Aa[kk],  b_packed(Pih, c * 4 + kk, lane), aR);
      aR = wmma16(Ahh[kk], b_packed(Phh, c * 4 + kk, lane), aR);
    }
#pragma unroll
    for (int kk = 0; kk < 4; ++kk) {  // z gate: iz + hz fused
      aZ = wmma16(Aa[kk],  b_packed(Pih, (8 + c) * 4 + kk, lane), aZ);
      aZ = wmma16(Ahh[kk], b_packed(Phh, (8 + c) * 4 + kk, lane), aZ);
    }
#pragma unroll
    for (int kk = 0; kk < 4; ++kk) {  // n gate halves (kept separate for r*hn)
      aN1 = wmma16(Aa[kk],  b_packed(Pih, (16 + c) * 4 + kk, lane), aN1);
      aN2 = wmma16(Ahh[kk], b_packed(Phh, (16 + c) * 4 + kk, lane), aN2);
    }
    const float brz = bih[j] + bhh[j];
    const float bzz = bih[HID + j] + bhh[HID + j];
    const float bni = bih[2 * HID + j];
    const float bnh = bhh[2 * HID + j];
    const float mu = bn_m[j], be = bn_b[j];
    const float iv = rsqrtf(bn_v[j] + BN_EPS) * bn_g[j];
#pragma unroll
    for (int v = 0; v < 8; ++v) {
      const size_t idx = (size_t)(tile * 16 + v + 8 * kh) * HID + j;
      const float hv = h[idx];
      const float r = sigm(aR[v] + brz);
      const float z = sigm(aZ[v] + bzz);
      const float nn = tanhf(aN1[v] + bni + r * (aN2[v] + bnh));
      const float hnew = (1.0f - z) * nn + z * hv;
      const float out = hv + (hnew - mu) * iv + be;  // residual + BN
      h[idx] = out;
      h16[idx] = (_Float16)out;
    }
  }
}

// ---------------------------------------------------------------------------
// Readout: segment mean/max via atomics, then relu([mean,max] @ ro_w + b)
// ---------------------------------------------------------------------------
__global__ void k_zero(float* __restrict__ p, int n) {
  int i = blockIdx.x * blockDim.x + threadIdx.x;
  if (i < n) p[i] = 0.0f;
}

__global__ void k_seg_init(float* __restrict__ gsum, unsigned* __restrict__ gmax,
                           int* __restrict__ gcnt) {
  int i = blockIdx.x * blockDim.x + threadIdx.x;
  if (i < NGRAPHS * HID) { gsum[i] = 0.0f; gmax[i] = 0u; }
  if (i < NGRAPHS) gcnt[i] = 0;
}

__global__ void k_seg_acc(const float* __restrict__ h, const int* __restrict__ batch,
                          float* __restrict__ gsum, unsigned* __restrict__ gmax,
                          int* __restrict__ gcnt) {
  int i = blockIdx.x * blockDim.x + threadIdx.x;
  if (i >= NNODES * HID) return;
  const int node = i >> 7, c = i & (HID - 1);
  const int g = batch[node];
  const float val = h[i];
  atomicAdd(&gsum[g * HID + c], val);
  atomicMax(&gmax[g * HID + c], fkey(val));
  if (c == 0) atomicAdd(&gcnt[g], 1);
}

__global__ void k_prep(const float* __restrict__ gsum, const unsigned* __restrict__ gmax,
                       const int* __restrict__ gcnt, float* __restrict__ zcat) {
  int i = blockIdx.x * blockDim.x + threadIdx.x;
  if (i >= NGRAPHS * HID) return;
  const int g = i >> 7, c = i & (HID - 1);
  const int cnt = gcnt[g];
  zcat[g * 2 * HID + c]       = gsum[i] / fmaxf((float)cnt, 1.0f);
  zcat[g * 2 * HID + HID + c] = (cnt > 0) ? funkey(gmax[i]) : 0.0f;
}

__global__ void k_readout(const float* __restrict__ zcat,
                          const _Float16* __restrict__ PW,  // 8 nt x 8 kt frags
                          const float* __restrict__ bias, float* __restrict__ out) {
  const int lane = threadIdx.x & 31, wid = threadIdx.x >> 5;
  const int tile = blockIdx.x * (blockDim.x >> 5) + wid;  // 16 graph tiles
  if (tile * 16 >= NGRAPHS) return;
  const int kh = lane >> 4, ln = lane & 15;
  const float* zrow = zcat + (size_t)(tile * 16 + ln) * (2 * HID);
  v16h A[8];
#pragma unroll
  for (int kk = 0; kk < 8; ++kk) A[kk] = a_frag_f(zrow, kk * 32, kh);
#pragma unroll
  for (int nt = 0; nt < 8; ++nt) {
    v8f acc = {};
#pragma unroll
    for (int kk = 0; kk < 8; ++kk)
      acc = wmma16(A[kk], b_packed(PW, nt * 8 + kk, lane), acc);
    const float bv = bias[nt * 16 + ln];
#pragma unroll
    for (int v = 0; v < 8; ++v)
      out[(size_t)(tile * 16 + v + 8 * kh) * HID + nt * 16 + ln] =
          fmaxf(acc[v] + bv, 0.0f);
  }
}

// ---------------------------------------------------------------------------
extern "C" void kernel_launch(void* const* d_in, const int* in_sizes, int n_in,
                              void* d_out, int out_size, void* d_ws, size_t ws_size,
                              hipStream_t stream) {
  const float* x       = (const float*)d_in[0];
  const int*   ei      = (const int*)d_in[1];
  const float* ea      = (const float*)d_in[2];
  const int*   batch   = (const int*)d_in[3];
  // d_in[4] = n_graphs scalar (compile-time constant here)
  const float* lin_w   = (const float*)d_in[5];
  const float* lin_b   = (const float*)d_in[6];
  const float* msg_w1  = (const float*)d_in[7];
  const float* msg_b1  = (const float*)d_in[8];
  const float* msg_w2  = (const float*)d_in[9];
  const float* msg_b2  = (const float*)d_in[10];
  const float* bn_g    = (const float*)d_in[11];
  const float* bn_b    = (const float*)d_in[12];
  const float* bn_m    = (const float*)d_in[13];
  const float* bn_v    = (const float*)d_in[14];
  const float* gru_wih = (const float*)d_in[15];
  const float* gru_whh = (const float*)d_in[16];
  const float* gru_bih = (const float*)d_in[17];
  const float* gru_bhh = (const float*)d_in[18];
  const float* ro_w    = (const float*)d_in[19];
  const float* ro_b    = (const float*)d_in[20];

  char* ws = (char*)d_ws;
  auto take = [&](size_t bytes) {
    char* p = ws;
    ws += (bytes + 255) & ~(size_t)255;
    return p;
  };
  float*    h    = (float*)take((size_t)NNODES * HID * 4);
  _Float16* h16  = (_Float16*)take((size_t)NNODES * HID * 2);
  float*    agg  = (float*)take((size_t)NNODES * HID * 4);
  float*    gsum = (float*)take((size_t)NGRAPHS * HID * 4);
  unsigned* gmax = (unsigned*)take((size_t)NGRAPHS * HID * 4);
  int*      gcnt = (int*)take((size_t)NGRAPHS * 4);
  float*    zcat = (float*)take((size_t)NGRAPHS * 2 * HID * 4);
  _Float16* p_lin  = (_Float16*)take((size_t)32 * FRAG * 2);
  _Float16* p_msg1 = (_Float16*)take((size_t)3 * 40 * FRAG * 2);
  _Float16* p_msg2 = (_Float16*)take((size_t)3 * 32 * FRAG * 2);
  _Float16* p_gih  = (_Float16*)take((size_t)3 * 96 * FRAG * 2);
  _Float16* p_ghh  = (_Float16*)take((size_t)3 * 96 * FRAG * 2);
  _Float16* p_ro   = (_Float16*)take((size_t)64 * FRAG * 2);

  const dim3 blk(256);  // 8 waves
  const int node_tiles = (NNODES + 15) / 16;  // 3125
  const int node_blocks = (node_tiles + 7) / 8;
  const int edge_tiles = NEDGES / 16;         // 50000
  const int edge_blocks =
      (edge_tiles + MSG_WAVES * MSG_TPW - 1) / (MSG_WAVES * MSG_TPW);  // 391

  // ---- pack weights to f16 fragment layout ----
  k_pack_km<<<(32 * FRAG + 255) / 256, blk, 0, stream>>>(lin_w, HID, 4, p_lin,
                                                         32 * FRAG);
  for (int l = 0; l < 3; ++l) {
    const float* w1 = msg_w1 + (size_t)l * (HID + EDIM) * HID;
    k_pack_km<<<(32 * FRAG + 255) / 256, blk, 0, stream>>>(
        w1, HID, 4, p_msg1 + (size_t)l * 40 * FRAG, 32 * FRAG);
    k_pack_edge<<<(8 * FRAG + 255) / 256, blk, 0, stream>>>(
        w1, p_msg1 + (size_t)l * 40 * FRAG + 32 * FRAG);
    k_pack_km<<<(32 * FRAG + 255) / 256, blk, 0, stream>>>(
        msg_w2 + (size_t)l * HID * HID, HID, 4, p_msg2 + (size_t)l * 32 * FRAG,
        32 * FRAG);
    k_pack_tr<<<(96 * FRAG + 255) / 256, blk, 0, stream>>>(
        gru_wih + (size_t)l * 3 * HID * HID, p_gih + (size_t)l * 96 * FRAG,
        96 * FRAG);
    k_pack_tr<<<(96 * FRAG + 255) / 256, blk, 0, stream>>>(
        gru_whh + (size_t)l * 3 * HID * HID, p_ghh + (size_t)l * 96 * FRAG,
        96 * FRAG);
  }
  k_pack_km<<<(64 * FRAG + 255) / 256, blk, 0, stream>>>(ro_w, HID, 8, p_ro,
                                                         64 * FRAG);

  // ---- network ----
  k_lin_in<<<node_blocks, blk, 0, stream>>>(x, p_lin, lin_b, h, h16);

  for (int l = 0; l < 3; ++l) {
    k_zero<<<(NNODES * HID + 255) / 256, blk, 0, stream>>>(agg, NNODES * HID);
    k_msg<<<edge_blocks, blk, 0, stream>>>(
        h16, ei, ea, p_msg1 + (size_t)l * 40 * FRAG, msg_b1 + l * HID,
        p_msg2 + (size_t)l * 32 * FRAG, msg_b2 + l * HID, agg);
    k_gru<<<node_blocks, blk, 0, stream>>>(
        h, h16, agg, p_gih + (size_t)l * 96 * FRAG, p_ghh + (size_t)l * 96 * FRAG,
        gru_bih + l * 3 * HID, gru_bhh + l * 3 * HID,
        bn_g + l * HID, bn_b + l * HID, bn_m + l * HID, bn_v + l * HID);
  }

  // ---- readout ----
  k_seg_init<<<(NGRAPHS * HID + 255) / 256, blk, 0, stream>>>(gsum, gmax, gcnt);
  k_seg_acc<<<(NNODES * HID + 255) / 256, blk, 0, stream>>>(h, batch, gsum, gmax,
                                                            gcnt);
  k_prep<<<(NGRAPHS * HID + 255) / 256, blk, 0, stream>>>(gsum, gmax, gcnt, zcat);
  k_readout<<<2, blk, 0, stream>>>(zcat, p_ro, ro_b, (float*)d_out);
}
